// CrossViewSwapAttention_42374147343018
// MI455X (gfx1250) — matrix operand order repaired and checked by reading, not compiled, not run on hardware
//
#include <hip/hip_runtime.h>
#include <hip/hip_bf16.h>

// CDNA5 (gfx1250) wave32 WMMA implementation of CrossViewSwapAttention.
// All matmuls -> v_wmma_f32_16x16x32_f16; LN/softmax/mean in f32 VALU.
// Attention: 8-wave workgroups stage K/V into LDS once via
// GLOBAL_LOAD_ASYNC_TO_LDS_B128 (ASYNCcnt path), 8 Q-tiles share them.

typedef __attribute__((ext_vector_type(16))) _Float16 v16h;
typedef __attribute__((ext_vector_type(8)))  _Float16 v8h;
typedef __attribute__((ext_vector_type(8)))  float    v8f;
typedef int v4i __attribute__((vector_size(16)));  // matches builtin param type

#define LN_EPS 1e-5f

#if defined(__has_builtin)
#if __has_builtin(__builtin_amdgcn_global_load_async_to_lds_b128) && \
    __has_builtin(__builtin_amdgcn_s_wait_asynccnt)
#define USE_ASYNC_LDS 1
#endif
#endif

#ifdef USE_ASYNC_LDS
// async 16B global -> LDS copy (per-lane), tracked by ASYNCcnt
__device__ __forceinline__ void async_g2l_b128(const _Float16* g, _Float16* l) {
  __builtin_amdgcn_global_load_async_to_lds_b128(
      (__attribute__((address_space(1))) v4i*)g,
      (__attribute__((address_space(3))) v4i*)l, 0, 0);
}
#endif

// ISA 7.12.2: 16-bit A-matrix 16x32 (MxK). Lane r=lane&15 holds row M=r.
// This code's order (not the ISA table order): element e holds K = e + 16*hi, one contiguous 16-half run
// starting at K = 16*hi (matches the contiguous B-fragment loads).
__device__ __forceinline__ v16h make_afrag(const _Float16* rowBase, int hi) {
  v8h lo = *reinterpret_cast<const v8h*>(rowBase + hi * 16);
  v8h hi8 = *reinterpret_cast<const v8h*>(rowBase + hi * 16 + 8);
  return __builtin_shufflevector(lo, hi8, 0, 1, 2, 3, 4, 5, 6, 7, 8, 9, 10, 11,
                                 12, 13, 14, 15);
}

__device__ __forceinline__ v8f wmma_f16(v16h a, v16h b, v8f c) {
  // (neg_a, A, neg_b, B, c_mod, C, reuse_a, reuse_b)
  return __builtin_amdgcn_wmma_f32_16x16x32_f16(false, a, false, b, (short)0, c,
                                                false, false);
}

// ---------------------------------------------------------------------------
// Weight pre-pass: f32 row-major [in=128][out=128] -> f16 transposed [out][in].
// ---------------------------------------------------------------------------
__global__ __launch_bounds__(256)
void weightT_kernel(const float* __restrict__ W, _Float16* __restrict__ WT) {
  int idx = blockIdx.x * 256 + threadIdx.x;  // 0..16383
  int k = idx >> 7, col = idx & 127;
  WT[col * 128 + k] = (_Float16)W[idx];
}

// ---------------------------------------------------------------------------
// Kernel A: gather (einops transpose) + LayerNorm + (x @ W + bias) -> f16.
// Block = 256 threads (8 waves), 16 rows x 128 cols.
// mode 0: Q gather (N=384, W2=8), row-major store
// mode 1: K gather (N=480, w2=10), row-major store
// mode 2: V gather (N=480, w2=10), transposed store VpT[l][m][dh][key]
// ---------------------------------------------------------------------------
__global__ __launch_bounds__(256)
void proj_ln_gemm_kernel(const float* __restrict__ xsrc,
                         const float* __restrict__ g,
                         const float* __restrict__ bln,
                         const _Float16* __restrict__ WT,  // [128 out][128 in]
                         const float* __restrict__ bias,   // [128]
                         _Float16* __restrict__ out,
                         int mode)
{
  __shared__ _Float16 xn[16][136];  // 136*2B keeps 16B alignment per chunk

  const int tid = threadIdx.x;
  const int rl  = tid >> 4;  // local row 0..15
  const int c   = tid & 15;  // 16 chunks of 8 elems
  const int r   = blockIdx.x * 16 + rl;

  // ---- gather source row ('b n x y w1 w2 d -> b (x y)(n w1 w2) d')
  long srcRow;
  if (mode == 0) {
    int l = r / 384, N = r % 384;
    int X = l >> 3, Y = l & 7;
    int n = N >> 6, rem = N & 63;
    int w1 = rem >> 3, w2 = rem & 7;
    srcRow = ((((long)(n * 8 + X)) * 8 + Y) * 8 + w1) * 8 + w2;
  } else {
    int l = r / 480, N = r % 480;
    int X = l >> 3, Y = l & 7;
    int n = N / 80, rem = N % 80;
    int w1 = rem / 10, w2 = rem % 10;
    srcRow = ((((long)(n * 8 + X)) * 8 + Y) * 8 + w1) * 10 + w2;
  }

  // ---- stage 1: LayerNorm over d=128 (16 lanes per row, 8 elems each)
  const float* src = xsrc + srcRow * 128 + c * 8;
  float4 a0 = reinterpret_cast<const float4*>(src)[0];
  float4 a1 = reinterpret_cast<const float4*>(src)[1];
  float v[8] = {a0.x, a0.y, a0.z, a0.w, a1.x, a1.y, a1.z, a1.w};
  float s = 0.f, s2 = 0.f;
#pragma unroll
  for (int j = 0; j < 8; ++j) { s += v[j]; s2 += v[j] * v[j]; }
#pragma unroll
  for (int msk = 1; msk < 16; msk <<= 1) {  // within 16-lane half (wave32)
    s  += __shfl_xor(s,  msk, 32);
    s2 += __shfl_xor(s2, msk, 32);
  }
  float mu   = s * (1.0f / 128.0f);
  float var  = s2 * (1.0f / 128.0f) - mu * mu;
  float rstd = rsqrtf(var + LN_EPS);
#pragma unroll
  for (int j = 0; j < 8; ++j) {
    int e = c * 8 + j;
    xn[rl][e] = (_Float16)((v[j] - mu) * rstd * g[e] + bln[e]);
  }
  __syncthreads();

  // ---- stage 2: xn(16x128) @ W(128x128), one 16-col tile per wave
  const int lane = tid & 31;
  const int ln16 = lane & 15;
  const int hi   = lane >> 4;
  const int col  = (tid >> 5) * 16 + ln16;

  v8f acc = {};
#pragma unroll
  for (int kk = 0; kk < 4; ++kk) {
    v16h afrag = make_afrag(&xn[ln16][kk * 32], hi);
    v16h bfrag = *reinterpret_cast<const v16h*>(WT + col * 128 + kk * 32 +
                                                hi * 16);
    acc = wmma_f16(afrag, bfrag, acc);
  }
  float bv = bias[col];

  if (mode == 2) {
    // transposed store: VpT[((l*4 + m)*32 + dh)*480 + key]; 16-row blocks
    // never straddle an l boundary (480 % 16 == 0) -> one v8h store per lane.
    int l = (blockIdx.x * 16) / 480;
    int keyBase = (blockIdx.x * 16) % 480;
    v8h st;
#pragma unroll
    for (int i = 0; i < 8; ++i) st[i] = (_Float16)(acc[i] + bv);
    _Float16* dst = out + ((long)((l * 4 + (col >> 5)) * 32 + (col & 31))) * 480
                    + keyBase + 8 * hi;
    *reinterpret_cast<v8h*>(dst) = st;
  } else {
    long orowBase = (long)blockIdx.x * 16;
#pragma unroll
    for (int i = 0; i < 8; ++i) {
      long orow = orowBase + i + 8 * hi;  // C/D layout: row = VGPR + 8*hi
      out[orow * 128 + col] = (_Float16)(acc[i] + bv);
    }
  }
}

// ---------------------------------------------------------------------------
// Kernel B: flash-style attention. Workgroup = 8 waves = 8 Q-tiles sharing
// one (l, head): K (480x32) and V^T (32x480) staged to LDS once, then each
// wave streams 15 key-pairs: 2 QK^T WMMAs + online softmax + 2 PV WMMAs.
// ---------------------------------------------------------------------------
__global__ __launch_bounds__(256)
void attn_kernel(const _Float16* __restrict__ Qp,   // [64*384][128]
                 const _Float16* __restrict__ Kp,   // [64*480][128]
                 const _Float16* __restrict__ VpT,  // [64][4][32][480]
                 const float* __restrict__ attn_scale,
                 _Float16* __restrict__ Ao)         // [64*384][128]
{
  __shared__ _Float16 Klds[480 * 32];      // [key][dh]   30720 B
  __shared__ _Float16 Vlds[32 * 480];      // [dh][key]   30720 B
  __shared__ _Float16 pTile[8][16][40];    // per-wave P staging (C->A relayout)

  const int l  = blockIdx.y;   // 0..63
  const int m  = blockIdx.z;   // 0..3
  const int tid  = threadIdx.x;
  const int wid  = tid >> 5;   // wave 0..7
  const int lane = tid & 31;
  const int ln16 = lane & 15;
  const int hi   = lane >> 4;
  const int qBase = (blockIdx.x * 8 + wid) * 16;
  const float sc = 0.17677669529663687f * attn_scale[0];  // 32^-0.5 * scale

  // ---- cooperative staging of K and V^T for this (l, m) into LDS ----------
  const _Float16* kSrc = Kp + (long)(l * 480) * 128 + m * 32;       // [480][32]
  const _Float16* vSrc = VpT + (long)((l * 4 + m) * 32) * 480;      // 15360 h
#ifdef USE_ASYNC_LDS
  for (int idx = tid; idx < 1920; idx += 256) {  // 1920 x 16B K chunks
    int row = idx >> 2, cc = idx & 3;
    async_g2l_b128(kSrc + (long)row * 128 + cc * 8, &Klds[row * 32 + cc * 8]);
  }
  for (int idx = tid; idx < 1920; idx += 256) {  // contiguous V^T block
    async_g2l_b128(vSrc + (long)idx * 8, &Vlds[idx * 8]);
  }
  __builtin_amdgcn_s_wait_asynccnt(0);
#else
  for (int idx = tid; idx < 1920; idx += 256) {
    int row = idx >> 2, cc = idx & 3;
    *reinterpret_cast<v8h*>(&Klds[row * 32 + cc * 8]) =
        *reinterpret_cast<const v8h*>(kSrc + (long)row * 128 + cc * 8);
  }
  for (int idx = tid; idx < 1920; idx += 256) {
    *reinterpret_cast<v8h*>(&Vlds[idx * 8]) =
        *reinterpret_cast<const v8h*>(vSrc + (long)idx * 8);
  }
#endif
  __syncthreads();

  // ---- Q A-fragment: rows = q rows, K dim = dh(32) -> one WMMA K-step -----
  v16h qf = make_afrag(Qp + ((long)(l * 384 + qBase + ln16)) * 128 + m * 32, hi);

  v8f o0 = {}, o1 = {};
  float mcur[8], lsum[8];
#pragma unroll
  for (int i = 0; i < 8; ++i) { mcur[i] = -3.0e38f; lsum[i] = 0.f; }

  for (int kp = 0; kp < 15; ++kp) {
    // S = Q K^T for two 16-key tiles; K B-frag = 16 contiguous halves in LDS
    v16h kf0 = *reinterpret_cast<const v16h*>(
        &Klds[(kp * 32 + ln16) * 32 + hi * 16]);
    v16h kf1 = *reinterpret_cast<const v16h*>(
        &Klds[(kp * 32 + 16 + ln16) * 32 + hi * 16]);
    v8f s0 = {}, s1 = {};
    s0 = wmma_f16(qf, kf0, s0);
    s1 = wmma_f16(qf, kf1, s1);

    // online softmax: one row-max reduction over both tiles
#pragma unroll
    for (int i = 0; i < 8; ++i) {
      float x0 = s0[i] * sc, x1 = s1[i] * sc;
      float mx = fmaxf(x0, x1);
#pragma unroll
      for (int msk = 1; msk < 16; msk <<= 1)
        mx = fmaxf(mx, __shfl_xor(mx, msk, 32));
      float nm = fmaxf(mcur[i], mx);
      float f  = __expf(mcur[i] - nm);
      mcur[i]  = nm;
      float p0 = __expf(x0 - nm);
      float p1 = __expf(x1 - nm);
      pTile[wid][i + 8 * hi][ln16]      = (_Float16)p0;
      pTile[wid][i + 8 * hi][16 + ln16] = (_Float16)p1;
      float rs = p0 + p1;
#pragma unroll
      for (int msk = 1; msk < 16; msk <<= 1) rs += __shfl_xor(rs, msk, 32);
      lsum[i] = lsum[i] * f + rs;
      o0[i] *= f;
      o1[i] *= f;
    }
    __syncthreads();  // uniform waves: cheap; orders P store -> A-frag load

    // P as A-fragment (16 q rows x 32 keys); V B-frags contiguous on key
    v16h pf  = make_afrag(&pTile[wid][ln16][0], hi);
    v16h vb0 = *reinterpret_cast<const v16h*>(
        &Vlds[ln16 * 480 + kp * 32 + hi * 16]);
    v16h vb1 = *reinterpret_cast<const v16h*>(
        &Vlds[(16 + ln16) * 480 + kp * 32 + hi * 16]);
    o0 = wmma_f16(pf, vb0, o0);
    o1 = wmma_f16(pf, vb1, o1);
  }

  // finalize: divide by row sums, write merged-head f16 output
#pragma unroll
  for (int i = 0; i < 8; ++i) {
    float inv = 1.0f / lsum[i];
    long orow = (long)(l * 384 + qBase + i + 8 * hi);
    Ao[orow * 128 + m * 32 + ln16]      = (_Float16)(o0[i] * inv);
    Ao[orow * 128 + m * 32 + 16 + ln16] = (_Float16)(o1[i] * inv);
  }
}

// ---------------------------------------------------------------------------
// Kernel C: mean over n commutes with @Wp -> average first (6x less GEMM),
// then (abar @ Wp) + bp + skip -> f32 out. Block = 16 rows x 128 cols.
// ---------------------------------------------------------------------------
__global__ __launch_bounds__(256)
void outproj_kernel(const _Float16* __restrict__ Ao,   // [64*384][128]
                    const _Float16* __restrict__ WpT,  // [128 out][128 in]
                    const float* __restrict__ bp,      // [128]
                    const float* __restrict__ skip,    // [4096][128]
                    float* __restrict__ out)           // [4096][128]
{
  __shared__ _Float16 xn[16][136];

  const int tid = threadIdx.x;
  const int rl  = tid >> 4;
  const int c   = tid & 15;
  const int r   = blockIdx.x * 16 + rl;  // output row in [0,4096)
  const int l   = r >> 6;                // X*8+Y
  const int ww  = r & 63;                // W1*8+W2

  float accm[8] = {};
#pragma unroll
  for (int n = 0; n < 6; ++n) {
    v8h a = *reinterpret_cast<const v8h*>(
        Ao + ((long)(l * 384 + n * 64 + ww)) * 128 + c * 8);
#pragma unroll
    for (int j = 0; j < 8; ++j) accm[j] += (float)a[j];
  }
#pragma unroll
  for (int j = 0; j < 8; ++j)
    xn[rl][c * 8 + j] = (_Float16)(accm[j] * (1.0f / 6.0f));
  __syncthreads();

  const int lane = tid & 31;
  const int ln16 = lane & 15;
  const int hi   = lane >> 4;
  const int col  = (tid >> 5) * 16 + ln16;

  v8f acc = {};
#pragma unroll
  for (int kk = 0; kk < 4; ++kk) {
    v16h afrag = make_afrag(&xn[ln16][kk * 32], hi);
    v16h bfrag = *reinterpret_cast<const v16h*>(WpT + col * 128 + kk * 32 +
                                                hi * 16);
    acc = wmma_f16(afrag, bfrag, acc);
  }
  float bv = bp[col];
  long orowBase = (long)blockIdx.x * 16;
#pragma unroll
  for (int i = 0; i < 8; ++i) {
    long off = (orowBase + i + 8 * hi) * 128 + col;
    out[off] = acc[i] + bv + skip[off];
  }
}

// ---------------------------------------------------------------------------
extern "C" void kernel_launch(void* const* d_in, const int* in_sizes, int n_in,
                              void* d_out, int out_size, void* d_ws,
                              size_t ws_size, hipStream_t stream) {
  (void)in_sizes; (void)n_in; (void)out_size; (void)ws_size;

  const float* q     = (const float*)d_in[0];
  const float* k     = (const float*)d_in[1];
  const float* v     = (const float*)d_in[2];
  const float* skip  = (const float*)d_in[3];
  const float* ascal = (const float*)d_in[4];
  const float* lnq_g = (const float*)d_in[5];
  const float* lnq_b = (const float*)d_in[6];
  const float* Wq    = (const float*)d_in[7];
  const float* bq    = (const float*)d_in[8];
  const float* lnk_g = (const float*)d_in[9];
  const float* lnk_b = (const float*)d_in[10];
  const float* Wk    = (const float*)d_in[11];
  const float* bk    = (const float*)d_in[12];
  const float* lnv_g = (const float*)d_in[13];
  const float* lnv_b = (const float*)d_in[14];
  const float* Wv    = (const float*)d_in[15];
  const float* bv    = (const float*)d_in[16];
  const float* Wp    = (const float*)d_in[17];
  const float* bp    = (const float*)d_in[18];

  // workspace layout (f16): Qp | Kp | VpT | Ao | WqT | WkT | WvT | WpT
  char* ws = (char*)d_ws;
  _Float16* Qp  = (_Float16*)(ws);                // 64*384*128*2 = 6,291,456 B
  _Float16* Kp  = (_Float16*)(ws + 6291456);      // 64*480*128*2 = 7,864,320 B
  _Float16* VpT = (_Float16*)(ws + 14155776);     // 7,864,320 B
  _Float16* Ao  = (_Float16*)(ws + 22020096);     // 6,291,456 B
  _Float16* WqT = (_Float16*)(ws + 28311552);     // 32,768 B each
  _Float16* WkT = (_Float16*)(ws + 28344320);
  _Float16* WvT = (_Float16*)(ws + 28377088);
  _Float16* WpT = (_Float16*)(ws + 28409856);

  weightT_kernel<<<64, 256, 0, stream>>>(Wq, WqT);
  weightT_kernel<<<64, 256, 0, stream>>>(Wk, WkT);
  weightT_kernel<<<64, 256, 0, stream>>>(Wv, WvT);
  weightT_kernel<<<64, 256, 0, stream>>>(Wp, WpT);

  proj_ln_gemm_kernel<<<1536, 256, 0, stream>>>(q, lnq_g, lnq_b, WqT, bq, Qp, 0);
  proj_ln_gemm_kernel<<<1920, 256, 0, stream>>>(k, lnk_g, lnk_b, WkT, bk, Kp, 1);
  proj_ln_gemm_kernel<<<1920, 256, 0, stream>>>(v, lnv_g, lnv_b, WvT, bv, VpT, 2);

  dim3 gA(3, 64, 4);
  attn_kernel<<<gA, 256, 0, stream>>>(Qp, Kp, VpT, ascal, Ao);

  outproj_kernel<<<256, 256, 0, stream>>>(Ao, WpT, bp, skip, (float*)d_out);
}
